// Attention_21655225106889
// MI455X (gfx1250) — matrix operand order repaired and checked by reading, not compiled, not run on hardware
//
#include <hip/hip_runtime.h>
#include <math.h>

#define DIM   384
#define HEADS 8
#define CH    48      // DIM / HEADS
#define HW    16384   // 128*128
#define BATCH 8

typedef __attribute__((ext_vector_type(16))) __bf16 v16bf;
typedef __attribute__((ext_vector_type(8)))  __bf16 v8bf;
typedef __attribute__((ext_vector_type(8)))  float  v8f;
typedef __attribute__((ext_vector_type(2)))  float  v2f;

// ---------------------------------------------------------------------------
// Convert the four 384x384 f32 weight matrices to bf16 (concatenated).
// ---------------------------------------------------------------------------
__global__ void cvt_w_kernel(const float* __restrict__ w0,
                             const float* __restrict__ w1,
                             const float* __restrict__ w2,
                             const float* __restrict__ wp,
                             __bf16* __restrict__ out) {
  int idx = blockIdx.x * 256 + threadIdx.x;           // 4*147456 total
  if (idx >= 4 * DIM * DIM) return;
  int which = idx / (DIM * DIM);
  int off   = idx % (DIM * DIM);
  const float* src = (which == 0) ? w0 : (which == 1) ? w1 : (which == 2) ? w2 : wp;
  out[idx] = (__bf16)src[off];
}

// ---------------------------------------------------------------------------
// [b][c][p] f32  ->  [b][p][c] bf16  (LDS 32x32 tiled transpose, both sides
// coalesced). grid = (HW/32, DIM/32, BATCH), block = 256.
// ---------------------------------------------------------------------------
__global__ void transpose_cvt_kernel(const float* __restrict__ in,
                                     __bf16* __restrict__ out) {
  __shared__ float tile[32][33];
  const int b  = blockIdx.z;
  const int ct = blockIdx.y;      // channel tile
  const int pt = blockIdx.x;      // pixel tile
  const int tx = threadIdx.x & 31;
  const int ty = threadIdx.x >> 5;    // 0..7
  const float* ip = in + ((size_t)b * DIM + ct * 32) * HW + pt * 32;
  #pragma unroll
  for (int r = ty; r < 32; r += 8)          // r = local c, tx = local p
    tile[r][tx] = ip[(size_t)r * HW + tx];
  __syncthreads();
  __bf16* op = out + ((size_t)b * HW + pt * 32) * DIM + ct * 32;
  #pragma unroll
  for (int r = ty; r < 32; r += 8)          // r = local p, tx = local c
    op[(size_t)r * DIM + tx] = (__bf16)tile[tx][r];
}

// ---------------------------------------------------------------------------
// 1x1 conv as bf16 WMMA GEMM:  Out[b][o][p] = sum_c W[o][c] * Xt[b][p][c].
// One wave computes a 32(M=o) x 64(N=p) tile; K = 384 in 12 steps of 32.
// Per K-step: 2 A-fragment + 4 B-fragment loads feed 8 WMMAs, and the
// fragments of step c0+32 are loaded BEFORE the WMMAs of step c0 issue
// (register double-buffering) so global loads overlap matrix math.
// A fragment (16x32 bf16): lane L (m = L&15, ka = L<16?0:8) holds
//   K = ka..ka+7 and ka+16..ka+23  -> two contiguous 16B loads of a W row.
// B fragment (32x16 bf16): lane L (n = L&15, kb = L<16?0:8) holds
//   K = kb..kb+7 and kb+16..kb+23 -> two contiguous 16B loads of an Xt row.
// C/D (f32): lane L VGPR r -> (M = r or r+8, N = L&15).
// grid = BATCH*12*256/4 = 6144 blocks of 128 threads (4 waves).
// ---------------------------------------------------------------------------
__global__ void __launch_bounds__(128)
gemm_bf16_kernel(const __bf16* __restrict__ W,    // [DIM][DIM] (O x C)
                 const __bf16* __restrict__ Xt,   // [BATCH][HW][DIM]
                 float* __restrict__ Out) {       // [BATCH][DIM][HW]
  const int lane = threadIdx.x & 31;
  const int wave = threadIdx.x >> 5;
  const long tile = (long)blockIdx.x * 4 + wave;
  const int tilesPerB = (DIM / 32) * (HW / 64);   // 3072
  const int b  = (int)(tile / tilesPerB);
  const int tr = (int)(tile % tilesPerB);
  const int o0 = (tr / (HW / 64)) * 32;
  const int p0 = (tr % (HW / 64)) * 64;

  const __bf16* Xb = Xt + (size_t)b * HW * DIM;
  const int m  = lane & 15;                 // A row within tile == B col
  const int ka = (lane < 16) ? 0 : 8;
  const int kb = (lane < 16) ? 0 : 8;

  const __bf16* wbase = W + (size_t)(o0 + m) * DIM + ka;   // A row tile 0
  const __bf16* xbase = Xb + (size_t)(p0 + m) * DIM + kb;  // B col base

  // fragment loaders -------------------------------------------------------
  #define LOAD_A(dst, c0, t) do {                                           \
    const __bf16* wr_ = wbase + (size_t)(t) * 16 * DIM + (c0);              \
    *((v8bf*)&(dst))     = *(const v8bf*)(wr_);                             \
    *((v8bf*)&(dst) + 1) = *(const v8bf*)(wr_ + 16);                        \
  } while (0)
  #define LOAD_B(dst, c0, j) do {                                           \
    const __bf16* xr_ = xbase + (size_t)(j) * 16 * DIM + (c0);              \
    *((v8bf*)&(dst))     = *(const v8bf*)(xr_);                             \
    *((v8bf*)&(dst) + 1) = *(const v8bf*)(xr_ + 16);                        \
  } while (0)

  v8f acc[8] = {};   // acc[t*4+j] : t = o-subtile (0..1), j = p-subtile (0..3)

  v16bf a0, a1, b0, b1, b2, b3;
  LOAD_A(a0, 0, 0); LOAD_A(a1, 0, 1);
  LOAD_B(b0, 0, 0); LOAD_B(b1, 0, 1); LOAD_B(b2, 0, 2); LOAD_B(b3, 0, 3);

  for (int c0 = 32; c0 < DIM; c0 += 32) {
    v16bf na0, na1, nb0, nb1, nb2, nb3;            // prefetch next K-step
    LOAD_A(na0, c0, 0); LOAD_A(na1, c0, 1);
    LOAD_B(nb0, c0, 0); LOAD_B(nb1, c0, 1); LOAD_B(nb2, c0, 2); LOAD_B(nb3, c0, 3);

    acc[0] = __builtin_amdgcn_wmma_f32_16x16x32_bf16(false, a0, false, b0, (short)0, acc[0], false, false);
    acc[1] = __builtin_amdgcn_wmma_f32_16x16x32_bf16(false, a0, false, b1, (short)0, acc[1], false, false);
    acc[2] = __builtin_amdgcn_wmma_f32_16x16x32_bf16(false, a0, false, b2, (short)0, acc[2], false, false);
    acc[3] = __builtin_amdgcn_wmma_f32_16x16x32_bf16(false, a0, false, b3, (short)0, acc[3], false, false);
    acc[4] = __builtin_amdgcn_wmma_f32_16x16x32_bf16(false, a1, false, b0, (short)0, acc[4], false, false);
    acc[5] = __builtin_amdgcn_wmma_f32_16x16x32_bf16(false, a1, false, b1, (short)0, acc[5], false, false);
    acc[6] = __builtin_amdgcn_wmma_f32_16x16x32_bf16(false, a1, false, b2, (short)0, acc[6], false, false);
    acc[7] = __builtin_amdgcn_wmma_f32_16x16x32_bf16(false, a1, false, b3, (short)0, acc[7], false, false);

    a0 = na0; a1 = na1; b0 = nb0; b1 = nb1; b2 = nb2; b3 = nb3;
  }
  // tail K-step
  acc[0] = __builtin_amdgcn_wmma_f32_16x16x32_bf16(false, a0, false, b0, (short)0, acc[0], false, false);
  acc[1] = __builtin_amdgcn_wmma_f32_16x16x32_bf16(false, a0, false, b1, (short)0, acc[1], false, false);
  acc[2] = __builtin_amdgcn_wmma_f32_16x16x32_bf16(false, a0, false, b2, (short)0, acc[2], false, false);
  acc[3] = __builtin_amdgcn_wmma_f32_16x16x32_bf16(false, a0, false, b3, (short)0, acc[3], false, false);
  acc[4] = __builtin_amdgcn_wmma_f32_16x16x32_bf16(false, a1, false, b0, (short)0, acc[4], false, false);
  acc[5] = __builtin_amdgcn_wmma_f32_16x16x32_bf16(false, a1, false, b1, (short)0, acc[5], false, false);
  acc[6] = __builtin_amdgcn_wmma_f32_16x16x32_bf16(false, a1, false, b2, (short)0, acc[6], false, false);
  acc[7] = __builtin_amdgcn_wmma_f32_16x16x32_bf16(false, a1, false, b3, (short)0, acc[7], false, false);
  #undef LOAD_A
  #undef LOAD_B

  #pragma unroll
  for (int t = 0; t < 2; ++t) {
    float* ob = Out + ((size_t)b * DIM + o0 + t * 16) * HW + p0 + m;
    #pragma unroll
    for (int r = 0; r < 8; ++r) {
      const size_t roff = (size_t)((lane < 16) ? r : r + 8) * HW;
      ob[roff]      = acc[t * 4 + 0][r];
      ob[roff + 16] = acc[t * 4 + 1][r];
      ob[roff + 32] = acc[t * 4 + 2][r];
      ob[roff + 48] = acc[t * 4 + 3][r];
    }
  }
}

// ---------------------------------------------------------------------------
// Depthwise 3x3, SAME zero padding. [b][c][128][128] f32 -> same shape.
// ---------------------------------------------------------------------------
__global__ void dwconv3x3_kernel(const float* __restrict__ in,
                                 const float* __restrict__ w,   // [DIM][9]
                                 float* __restrict__ out) {
  size_t idx = (size_t)blockIdx.x * 256 + threadIdx.x;   // BATCH*DIM*HW
  const int x  = (int)(idx & 127);
  const int y  = (int)((idx >> 7) & 127);
  const size_t bc = idx >> 14;
  const int c = (int)(bc % DIM);
  const float* wp = w + c * 9;
  const float* ip = in + (bc << 14);
  float s = 0.f;
  #pragma unroll
  for (int dy = -1; dy <= 1; ++dy) {
    int yy = y + dy;
    if ((unsigned)yy >= 128u) continue;
    #pragma unroll
    for (int dx = -1; dx <= 1; ++dx) {
      int xx = x + dx;
      if ((unsigned)xx >= 128u) continue;
      s += ip[yy * 128 + xx] * wp[(dy + 1) * 3 + (dx + 1)];
    }
  }
  out[idx] = s;
}

// ---------------------------------------------------------------------------
// Per-channel L2 norm over the 16384 pixels: dn[row] = max(||row||, eps).
// One block per (b,c) row.
// ---------------------------------------------------------------------------
__global__ void rowsumsq_kernel(const float* __restrict__ in,
                                float* __restrict__ dn) {
  __shared__ float red[256];
  const size_t row = blockIdx.x;
  const float* p = in + (row << 14);
  float s = 0.f;
  for (int i = threadIdx.x; i < HW; i += 256) { float v = p[i]; s += v * v; }
  red[threadIdx.x] = s;
  __syncthreads();
  for (int off = 128; off; off >>= 1) {
    if (threadIdx.x < off) red[threadIdx.x] += red[threadIdx.x + off];
    __syncthreads();
  }
  if (threadIdx.x == 0) dn[row] = fmaxf(sqrtf(red[0]), 1e-12f);
}

__global__ void zerof_kernel(float* p, int n) {
  int i = blockIdx.x * 256 + threadIdx.x;
  if (i < n) p[i] = 0.f;
}

// ---------------------------------------------------------------------------
// Channel gram S[bh][c][d] = sum_p q[c][p]*k[d][p], f32 WMMA 16x16x4.
// One wave per (b,h,ksplit); K chunk = 512; 9 accumulator tiles (48x48).
// A frag (16x4 f32): lane L: m=L&15, kbase=(L<16)?0:2, a[r]=q[m][k+kbase+r].
// B frag (4x16 f32): same per-lane pattern with n=L&15 on k rows.
// grid = BATCH*HEADS*32 blocks of 32 threads.
// ---------------------------------------------------------------------------
__global__ void attn_qk_kernel(const float* __restrict__ q,
                               const float* __restrict__ k,
                               float* __restrict__ S) {
  const int lane = threadIdx.x & 31;
  const int bh = blockIdx.x >> 5;
  const int ks = blockIdx.x & 31;
  const float* qb = q + (size_t)bh * CH * HW;
  const float* kb = k + (size_t)bh * CH * HW;
  const int mn = lane & 15;
  const int kbase = (lane < 16) ? 0 : 2;
  v8f acc[9] = {};
  const int k0 = ks * (HW / 32);
  const int k1 = k0 + (HW / 32);

  v2f af[3], bf[3];
  #pragma unroll
  for (int i = 0; i < 3; ++i) {
    af[i] = *(const v2f*)(qb + (size_t)(i * 16 + mn) * HW + k0 + kbase);
    bf[i] = *(const v2f*)(kb + (size_t)(i * 16 + mn) * HW + k0 + kbase);
  }
  for (int kk = k0; kk < k1; kk += 4) {
    v2f naf[3], nbf[3];
    if (kk + 4 < k1) {                       // prefetch next K-step
      #pragma unroll
      for (int i = 0; i < 3; ++i) {
        naf[i] = *(const v2f*)(qb + (size_t)(i * 16 + mn) * HW + kk + 4 + kbase);
        nbf[i] = *(const v2f*)(kb + (size_t)(i * 16 + mn) * HW + kk + 4 + kbase);
      }
    }
    #pragma unroll
    for (int i = 0; i < 3; ++i)
      #pragma unroll
      for (int j = 0; j < 3; ++j)
        acc[i * 3 + j] = __builtin_amdgcn_wmma_f32_16x16x4_f32(
            false, af[i], false, bf[j], (short)0, acc[i * 3 + j], false, false);
    #pragma unroll
    for (int i = 0; i < 3; ++i) { af[i] = naf[i]; bf[i] = nbf[i]; }
  }
  float* Sp = S + (size_t)bh * CH * CH;
  #pragma unroll
  for (int i = 0; i < 3; ++i)
    #pragma unroll
    for (int j = 0; j < 3; ++j)
      #pragma unroll
      for (int r = 0; r < 8; ++r) {
        const int row = i * 16 + ((lane < 16) ? r : r + 8);
        const int col = j * 16 + mn;
        atomicAdd(&Sp[row * CH + col], acc[i * 3 + j][r]);
      }
}

// ---------------------------------------------------------------------------
// Scaled softmax over d: P[c][d] = softmax_d( S[c][d]/(nq_c*nk_d) * temp_h ).
// One thread per (b,h,c) row of 48. In-place on S.
// ---------------------------------------------------------------------------
__global__ void softmax48_kernel(float* __restrict__ S,
                                 const float* __restrict__ nq,
                                 const float* __restrict__ nk,
                                 const float* __restrict__ temp) {
  int idx = blockIdx.x * 256 + threadIdx.x;    // BATCH*HEADS*CH = 3072
  if (idx >= BATCH * HEADS * CH) return;
  const int c  = idx % CH;
  const int bh = idx / CH;
  const int h  = bh % HEADS;
  const int b  = bh / HEADS;
  float* row = S + (size_t)bh * CH * CH + c * CH;
  const float* dq = nq + b * DIM + h * CH;
  const float* dk = nk + b * DIM + h * CH;
  const float sc = temp[h] / dq[c];
  float vals[CH];
  float mx = -3.0e38f;
  #pragma unroll
  for (int d = 0; d < CH; ++d) {
    vals[d] = row[d] * sc / dk[d];
    mx = fmaxf(mx, vals[d]);
  }
  float sum = 0.f;
  #pragma unroll
  for (int d = 0; d < CH; ++d) { vals[d] = expf(vals[d] - mx); sum += vals[d]; }
  const float inv = 1.f / sum;
  #pragma unroll
  for (int d = 0; d < CH; ++d) row[d] = vals[d] * inv;
}

// ---------------------------------------------------------------------------
// out[bh][c][p] = sum_d P[c][d] * v[bh][d][p]. Memory-bound: each thread
// holds v[*][p] in 48 registers, P in LDS. grid = BATCH*HEADS*64, block=256.
// ---------------------------------------------------------------------------
__global__ void attnv_kernel(const float* __restrict__ P,
                             const float* __restrict__ v,
                             float* __restrict__ out) {
  __shared__ float Ps[CH * CH];
  const int bh = blockIdx.x >> 6;
  const int pt = blockIdx.x & 63;
  for (int i = threadIdx.x; i < CH * CH; i += 256)
    Ps[i] = P[(size_t)bh * CH * CH + i];
  __syncthreads();
  const int p = pt * 256 + threadIdx.x;
  const float* vb = v + (size_t)bh * CH * HW;
  float vr[CH];
  #pragma unroll
  for (int d = 0; d < CH; ++d) vr[d] = vb[(size_t)d * HW + p];
  float* ob = out + (size_t)bh * CH * HW;
  for (int c = 0; c < CH; ++c) {
    float s = 0.f;
    #pragma unroll
    for (int d = 0; d < CH; ++d) s += Ps[c * CH + d] * vr[d];
    ob[(size_t)c * HW + p] = s;
  }
}

// ---------------------------------------------------------------------------
extern "C" void kernel_launch(void* const* d_in, const int* in_sizes, int n_in,
                              void* d_out, int out_size, void* d_ws, size_t ws_size,
                              hipStream_t stream) {
  const float* x     = (const float*)d_in[0];
  const float* w0    = (const float*)d_in[1];
  const float* w1    = (const float*)d_in[2];
  const float* w2    = (const float*)d_in[3];
  const float* dw[3] = {(const float*)d_in[4], (const float*)d_in[5], (const float*)d_in[6]};
  const float* wproj = (const float*)d_in[7];
  const float* temp  = (const float*)d_in[8];

  const size_t NELEM = (size_t)BATCH * DIM * HW;       // 50,331,648
  char* ws = (char*)d_ws;
  __bf16* xbt = (__bf16*)ws;                           // [B][HW][DIM] bf16
  const size_t XBT_BYTES = NELEM * 2;                  // 100,663,296
  __bf16* wb = (__bf16*)(ws + XBT_BYTES);              // 4 x [DIM][DIM] bf16
  const size_t WB_BYTES = (size_t)4 * DIM * DIM * 2;   // 1,179,648
  float* fbase = (float*)(ws + XBT_BYTES + WB_BYTES);
  float* tbuf = fbase;                                 // conv out / attn out (reused)
  float* qbuf = tbuf + NELEM;
  float* kbuf = qbuf + NELEM;
  float* vbuf = kbuf + NELEM;
  float* nq   = vbuf + NELEM;                          // [B*DIM]
  float* nk   = nq + BATCH * DIM;
  float* Sbuf = nk + BATCH * DIM;                      // [B*HEADS][48][48]
  float* qkv[3] = {qbuf, kbuf, vbuf};

  const dim3 tblk(256);

  // weights -> bf16
  cvt_w_kernel<<<(4 * DIM * DIM + 255) / 256, tblk, 0, stream>>>(w0, w1, w2, wproj, wb);
  // x -> transposed bf16 activation matrix (fits in 192MB L2 for q/k/v GEMMs)
  transpose_cvt_kernel<<<dim3(HW / 32, DIM / 32, BATCH), tblk, 0, stream>>>(x, xbt);

  // q / k / v branches: WMMA GEMM then depthwise 3x3
  const int gemmBlocks = (BATCH * (DIM / 32) * (HW / 64)) / 4;   // 6144
  const int dwBlocks   = (int)(NELEM / 256);                     // 196608
  for (int i = 0; i < 3; ++i) {
    gemm_bf16_kernel<<<gemmBlocks, 128, 0, stream>>>(wb + (size_t)i * DIM * DIM, xbt, tbuf);
    dwconv3x3_kernel<<<dwBlocks, tblk, 0, stream>>>(tbuf, dw[i], qkv[i]);
  }

  // L2 norms of q/k channel rows
  rowsumsq_kernel<<<BATCH * DIM, tblk, 0, stream>>>(qbuf, nq);
  rowsumsq_kernel<<<BATCH * DIM, tblk, 0, stream>>>(kbuf, nk);

  // channel gram via f32 WMMA + atomics, then scaled softmax
  const int Sn = BATCH * HEADS * CH * CH;                        // 147456
  zerof_kernel<<<(Sn + 255) / 256, tblk, 0, stream>>>(Sbuf, Sn);
  attn_qk_kernel<<<BATCH * HEADS * 32, 32, 0, stream>>>(qbuf, kbuf, Sbuf);
  softmax48_kernel<<<(BATCH * HEADS * CH + 255) / 256, tblk, 0, stream>>>(Sbuf, nq, nk, temp);

  // attn @ v  -> tbuf (reuse)
  attnv_kernel<<<BATCH * HEADS * 64, tblk, 0, stream>>>(Sbuf, vbuf, tbuf);

  // final projection: transpose+bf16 (reuse xbt) then WMMA GEMM into d_out
  transpose_cvt_kernel<<<dim3(HW / 32, DIM / 32, BATCH), tblk, 0, stream>>>(tbuf, xbt);
  gemm_bf16_kernel<<<gemmBlocks, 128, 0, stream>>>(wb + (size_t)3 * DIM * DIM, xbt, (float*)d_out);
}
